// SFSRec_26156350832984
// MI455X (gfx1250) — matrix operand order repaired and checked by reading, not compiled, not run on hardware
//
#include <hip/hip_runtime.h>
#include <hip/hip_bf16.h>
#include <stdint.h>

// ---------------------------------------------------------------------------
// SFSRec forward on gfx1250 (MI455X).  Memory-bound workload: all activations
// kept in _Float16 (exact for spikes / small-int residual stream), GEMMs via
// v_wmma_f32_16x16x32_f16 with async-to-LDS double-buffered staging.
// ---------------------------------------------------------------------------

typedef _Float16 h16;
typedef __attribute__((ext_vector_type(16))) _Float16 v16h;
typedef __attribute__((ext_vector_type(8)))  _Float16 v8h;
typedef __attribute__((ext_vector_type(8)))  float    v8f;

#define T_STEPS 4
#define BATCH   256
#define SEQ     200
#define HID     256
#define HDIM    128
#define BL      (BATCH*SEQ)        /* 51200  */
#define NTOK    (T_STEPS*BL)       /* 204800 */
#define SCALE_QK 0.125f

union F16x16 { v16h v; v8h h[2]; };

// Async global->LDS path (gfx1250): builtin signature is
//   (int4 AS1* src, int4 AS3* dst, imm offset, imm cpol)
#if defined(__AMDGCN__) && __has_builtin(__builtin_amdgcn_global_load_async_to_lds_b128) && __has_builtin(__builtin_amdgcn_s_wait_asynccnt)
#define ASYNC_OK 1
typedef int i32x4 __attribute__((ext_vector_type(4)));
typedef __attribute__((address_space(1))) i32x4 as1_i32x4;
typedef __attribute__((address_space(3))) i32x4 as3_i32x4;
#else
#define ASYNC_OK 0
#endif

// ---------------------------------------------------------------------------
// Weight f32 -> f16 conversion
// ---------------------------------------------------------------------------
__global__ void conv_f32_f16(const float* __restrict__ src, h16* __restrict__ dst, int n) {
    int i = blockIdx.x * 256 + threadIdx.x;
    if (i < n) dst[i] = (h16)src[i];
}

// ---------------------------------------------------------------------------
// GEMM: G[N,256] = X[N,256] (f16) * W[256,256]^T (f16), f32 out.
// Block tile 128 (rows) x 64 (cols), 8 waves, each wave 2x2 WMMA 16x16 tiles.
// Double-buffered LDS staging via GLOBAL_LOAD_ASYNC_TO_LDS_B128 (ASYNCcnt).
// ---------------------------------------------------------------------------
__global__ __launch_bounds__(256) void gemm_xw_f16(
    const h16* __restrict__ Xp, const h16* __restrict__ Wp, float* __restrict__ Gp)
{
    __shared__ __attribute__((aligned(32))) h16 Xs[2][128][32];
    __shared__ __attribute__((aligned(32))) h16 Ws[2][64][32];

    const int tid  = threadIdx.x;
    const int lane = tid & 31;
    const int wv   = tid >> 5;       // wave 0..7
    const int wr   = wv >> 1;        // 32-row group 0..3
    const int wc   = wv & 1;         // 32-col group 0..1
    const size_t row0 = (size_t)blockIdx.x * 128;
    const int    col0 = blockIdx.y * 64;

    const int lr   = lane & 15;
    const int asel = (lane < 16) ? 0 : 8;    // A: K-halves select (ISA 16-bit A layout)
    const int bsel = (lane < 16) ? 0 : 8;    // B: K-row select    (ISA 16-bit B layout)

    // per-thread staging addresses (this thread moves 32B of X, 16B of W)
    const h16* xg = Xp + (row0 + (size_t)(tid >> 1)) * 256 + (tid & 1) * 16;
    const h16* wg = Wp + (size_t)(col0 + (tid >> 2)) * 256 + (tid & 3) * 8;

    auto stage = [&](int buf, int k0) {
        h16* xs = &Xs[buf][tid >> 1][(tid & 1) * 16];
        h16* ws = &Ws[buf][tid >> 2][(tid & 3) * 8];
#if ASYNC_OK
        __builtin_amdgcn_global_load_async_to_lds_b128((as1_i32x4*)(xg + k0),     (as3_i32x4*)xs,       0, 0);
        __builtin_amdgcn_global_load_async_to_lds_b128((as1_i32x4*)(xg + k0 + 8), (as3_i32x4*)(xs + 8), 0, 0);
        __builtin_amdgcn_global_load_async_to_lds_b128((as1_i32x4*)(wg + k0),     (as3_i32x4*)ws,       0, 0);
#else
        const uint4* g = (const uint4*)(xg + k0);
        uint4* s = (uint4*)xs;
        s[0] = g[0]; s[1] = g[1];
        *(uint4*)ws = *(const uint4*)(wg + k0);
#endif
    };

    v8f zero = {};
    v8f acc[2][2];
#pragma unroll
    for (int i = 0; i < 2; i++)
#pragma unroll
        for (int j = 0; j < 2; j++) acc[i][j] = zero;

    stage(0, 0);                                   // prologue: tile 0 in flight

#pragma unroll
    for (int k0 = 0; k0 < 256; k0 += 32) {
        const int cur = (k0 >> 5) & 1;
        if (k0 + 32 < 256) {
            stage(cur ^ 1, k0 + 32);               // DMA next tile into other buffer
#if ASYNC_OK
            __builtin_amdgcn_s_wait_asynccnt(3);   // current tile landed, next in flight
#endif
        } else {
#if ASYNC_OK
            __builtin_amdgcn_s_wait_asynccnt(0);   // last tile landed
#endif
        }
        __syncthreads();                           // whole tile visible to all waves

        v16h a[2], b[2];
#pragma unroll
        for (int i = 0; i < 2; i++) {
            F16x16 t;
            t.h[0] = *(const v8h*)&Xs[cur][wr * 32 + i * 16 + lr][asel];
            t.h[1] = *(const v8h*)&Xs[cur][wr * 32 + i * 16 + lr][asel + 16];
            a[i] = t.v;
        }
#pragma unroll
        for (int j = 0; j < 2; j++) {
            F16x16 t;
            t.h[0] = *(const v8h*)&Ws[cur][wc * 32 + j * 16 + lr][bsel];
            t.h[1] = *(const v8h*)&Ws[cur][wc * 32 + j * 16 + lr][bsel + 16];
            b[j] = t.v;
        }
#pragma unroll
        for (int i = 0; i < 2; i++)
#pragma unroll
            for (int j = 0; j < 2; j++)
                acc[i][j] = __builtin_amdgcn_wmma_f32_16x16x32_f16(
                    false, a[i], false, b[j], (short)0, acc[i][j], false, false);
        __syncthreads();                           // readers done before buffer re-targeted
    }

    const int rb = (lane < 16) ? 0 : 8;   // C layout: (vgpr e, lane) = C[e+rb][lane&15]
#pragma unroll
    for (int i = 0; i < 2; i++)
#pragma unroll
        for (int j = 0; j < 2; j++)
#pragma unroll
            for (int e = 0; e < 8; e++) {
                size_t r = row0 + wr * 32 + i * 16 + e + rb;
                int    c = col0 + wc * 32 + j * 16 + lr;
                Gp[r * 256 + c] = acc[i][j][e];
            }
}

// ---------------------------------------------------------------------------
// Fused (bias +) LayerNorm + 4-step LIF.  One block per (b,l), thread = feature.
// Optionally accumulates spikes into residual stream Xres.
// ---------------------------------------------------------------------------
__global__ __launch_bounds__(256) void ln_lif_kernel(
    const float* __restrict__ G, const float* __restrict__ bias,
    const float* __restrict__ gamma, const float* __restrict__ beta,
    float eps, float vth, h16* __restrict__ S, h16* __restrict__ Xres)
{
    const int bl = blockIdx.x, o = threadIdx.x;
    __shared__ float s1[256], s2[256];
    const float bo = bias ? bias[o] : 0.0f;
    const float ga = gamma[o], be = beta[o];
    float y[T_STEPS];
    for (int t = 0; t < T_STEPS; t++) {
        float g = G[((size_t)t * BL + bl) * 256 + o] + bo;
        s1[o] = g; s2[o] = g * g;
        __syncthreads();
        for (int st = 128; st > 0; st >>= 1) {
            if (o < st) { s1[o] += s1[o + st]; s2[o] += s2[o + st]; }
            __syncthreads();
        }
        float mean = s1[0] * (1.0f / 256.0f);
        float var  = s2[0] * (1.0f / 256.0f) - mean * mean;
        __syncthreads();
        y[t] = (g - mean) * rsqrtf(var + eps) * ga + be;
    }
    float v = 0.0f;
    for (int t = 0; t < T_STEPS; t++) {
        v = 0.5f * (v + y[t]);                 // v += (x - v)/tau, tau = 2
        float sp = (v >= vth) ? 1.0f : 0.0f;
        v *= (1.0f - sp);                      // hard reset
        size_t idx = ((size_t)t * BL + bl) * 256 + o;
        S[idx] = (h16)sp;
        if (Xres) Xres[idx] = (h16)((float)Xres[idx] + sp);
    }
}

// ---------------------------------------------------------------------------
// Plain 4-step LIF (attn_lif, v_th = 0.5), f32 in -> f16 spikes out.
// ---------------------------------------------------------------------------
__global__ __launch_bounds__(256) void lif_kernel(
    const float* __restrict__ G, h16* __restrict__ S, float vth)
{
    size_t idx = (size_t)blockIdx.x * 256 + threadIdx.x;
    float v = 0.0f;
    for (int t = 0; t < T_STEPS; t++) {
        float x = G[(size_t)t * BL * 256 + idx];
        v = 0.5f * (v + x);
        float sp = (v >= vth) ? 1.0f : 0.0f;
        v *= (1.0f - sp);
        S[(size_t)t * BL * 256 + idx] = (h16)sp;
    }
}

// ---------------------------------------------------------------------------
// Embedding gather + pos add + LayerNorm(eps 1e-12) + rpe-LIF (v_th=1).
// x is identical across T, so one block computes all 4 spike planes.
// ---------------------------------------------------------------------------
__global__ __launch_bounds__(256) void embed_kernel(
    const int* __restrict__ seq, const float* __restrict__ emb,
    const float* __restrict__ pos, const float* __restrict__ gamma,
    const float* __restrict__ beta, h16* __restrict__ X)
{
    const int bl = blockIdx.x, o = threadIdx.x, l = bl % SEQ;
    __shared__ float s1[256], s2[256];
    const int it = seq[bl];
    float x = emb[(size_t)it * 256 + o] + pos[(size_t)l * 256 + o];
    s1[o] = x; s2[o] = x * x;
    __syncthreads();
    for (int st = 128; st > 0; st >>= 1) {
        if (o < st) { s1[o] += s1[o + st]; s2[o] += s2[o + st]; }
        __syncthreads();
    }
    float mean = s1[0] * (1.0f / 256.0f);
    float var  = s2[0] * (1.0f / 256.0f) - mean * mean;
    float y = (x - mean) * rsqrtf(var + 1e-12f) * gamma[o] + beta[o];
    float v = 0.0f;
    for (int t = 0; t < T_STEPS; t++) {
        v = 0.5f * (v + y);
        float sp = (v >= 1.0f) ? 1.0f : 0.0f;
        v *= (1.0f - sp);
        X[((size_t)t * BL + bl) * 256 + o] = (h16)sp;
    }
}

// ---------------------------------------------------------------------------
// Attention (no softmax): Y = (Q K^T * 0.125) V per (t,b,head).
// grid = (13 i-tiles, 2 heads, T*B).  Stage 1: attn strip [16x224] in LDS via
// WMMA; cols >= 200 zeroed.  Stage 2: attn * V via WMMA, V slab transposed
// through LDS per 32-wide K-step, wave w owns d-tile w.
// ---------------------------------------------------------------------------
__global__ __launch_bounds__(256) void attention_kernel(
    const h16* __restrict__ Sq, const h16* __restrict__ Sk,
    const h16* __restrict__ Sv, float* __restrict__ G)
{
    __shared__ __attribute__((aligned(32))) h16 As[16][224];
    __shared__ __attribute__((aligned(32))) h16 Vt[128][40];   // 80B row stride, 16B aligned

    const int tid  = threadIdx.x, lane = tid & 31, wv = tid >> 5;
    const int itile = blockIdx.x;                // 0..12
    const int hoff  = blockIdx.y * HDIM;         // head offset in feature dim
    const size_t rowbase = (size_t)blockIdx.z * SEQ;

    const int lr   = lane & 15;
    const int asel = (lane < 16) ? 0 : 8;
    const int bsel = (lane < 16) ? 0 : 8;
    const int rb   = (lane < 16) ? 0 : 8;
    v8f zero = {};

    // ---- stage 1: As[i][j] = 0.125 * sum_d Q[i0+i,d] * K[j0+j,d]
    const h16* Qb = Sq + (rowbase + itile * 16) * 256 + hoff;
    for (int jt = wv; jt < 13; jt += 8) {
        v8f acc = zero;
        const h16* Kb = Sk + (rowbase + jt * 16) * 256 + hoff;
#pragma unroll
        for (int d0 = 0; d0 < HDIM; d0 += 32) {
            F16x16 ta, tbf;
            ta.h[0]  = *(const v8h*)(Qb + (size_t)lr * 256 + d0 + asel);
            ta.h[1]  = *(const v8h*)(Qb + (size_t)lr * 256 + d0 + asel + 16);
            tbf.h[0] = *(const v8h*)(Kb + (size_t)lr * 256 + d0 + bsel);
            tbf.h[1] = *(const v8h*)(Kb + (size_t)lr * 256 + d0 + bsel + 16);
            acc = __builtin_amdgcn_wmma_f32_16x16x32_f16(
                false, ta.v, false, tbf.v, (short)0, acc, false, false);
        }
#pragma unroll
        for (int e = 0; e < 8; e++)
            As[e + rb][jt * 16 + lr] = (h16)(acc[e] * SCALE_QK);
    }
    __syncthreads();
    // zero attn columns j >= 200 (garbage K rows / padding)
    for (int idx = tid; idx < 16 * 24; idx += 256)
        As[idx / 24][200 + idx % 24] = (h16)0.0f;
    __syncthreads();

    // ---- stage 2: Y[i, d] = sum_j As[i][j] * V[j][d]
    v8f acc2 = zero;
    for (int j0 = 0; j0 < 224; j0 += 32) {
        {   // stage V slab transposed: Vt[d][jj] = V[j0+jj][d], zeros past SEQ
            const int jj = tid >> 3;           // 0..31
            const int c8 = (tid & 7) * 16;     // 0..112
            const int j  = j0 + jj;
            if (j < SEQ) {
                const h16* vp = Sv + (rowbase + j) * 256 + hoff + c8;
#pragma unroll
                for (int e = 0; e < 16; e++) Vt[c8 + e][jj] = vp[e];
            } else {
#pragma unroll
                for (int e = 0; e < 16; e++) Vt[c8 + e][jj] = (h16)0.0f;
            }
        }
        __syncthreads();
        F16x16 ta, tbf;
        ta.h[0]  = *(const v8h*)&As[lr][j0 + asel];
        ta.h[1]  = *(const v8h*)&As[lr][j0 + asel + 16];
        tbf.h[0] = *(const v8h*)&Vt[wv * 16 + lr][bsel];
        tbf.h[1] = *(const v8h*)&Vt[wv * 16 + lr][bsel + 16];
        acc2 = __builtin_amdgcn_wmma_f32_16x16x32_f16(
            false, ta.v, false, tbf.v, (short)0, acc2, false, false);
        __syncthreads();
    }
#pragma unroll
    for (int e = 0; e < 8; e++) {
        int li = itile * 16 + e + rb;
        if (li < SEQ)
            G[(rowbase + li) * 256 + hoff + wv * 16 + lr] = acc2[e];
    }
}

// ---------------------------------------------------------------------------
// Head: gather last valid position, mean over T, [256]x[256,256]^T + bias.
// ---------------------------------------------------------------------------
__global__ __launch_bounds__(256) void head_kernel(
    const h16* __restrict__ X, const int* __restrict__ lens,
    const float* __restrict__ Wh, const float* __restrict__ bh,
    float* __restrict__ out)
{
    __shared__ float xm[256];
    const int b = blockIdx.x, o = threadIdx.x;
    const int r = lens[b] - 1;
    const size_t base = ((size_t)b * SEQ + r) * 256;
    float s = 0.0f;
    for (int t = 0; t < T_STEPS; t++) s += (float)X[(size_t)t * BL * 256 + base + o];
    xm[o] = s * 0.25f;
    __syncthreads();
    float acc = bh[o];
    const float* wr = Wh + (size_t)o * 256;
    for (int h = 0; h < 256; h++) acc = fmaf(xm[h], wr[h], acc);
    out[(size_t)b * 256 + o] = acc;
}

// ---------------------------------------------------------------------------
// Host orchestration.  Input flattening assumed = JAX pytree order (dict keys
// sorted, 'W' < 'b' < 'bias' < 'g' in ASCII):
//   0 item_seq, 1 item_seq_len,
//   per block (base = 2 + 21*blk):
//     +0 fc1_W  +1 fc1_b(LN beta) +2 fc1_bias +3 fc1_g
//     +4 fc2_W  +5 fc2_b          +6 fc2_bias +7 fc2_g
//     +8 k_W    +9 k_b            +10 k_g
//     +11 proj_W +12 proj_b(LN beta) +13 proj_bias +14 proj_g
//     +15 q_W   +16 q_b           +17 q_g
//     +18 v_W   +19 v_b           +20 v_g
//   44 head_W, 45 head_b, 46 item_emb, 47 ln_b, 48 ln_g, 49 pos_emb
// ---------------------------------------------------------------------------
extern "C" void kernel_launch(void* const* d_in, const int* in_sizes, int n_in,
                              void* d_out, int out_size, void* d_ws, size_t ws_size,
                              hipStream_t stream)
{
    (void)in_sizes; (void)n_in; (void)out_size; (void)ws_size;
    const int* item_seq = (const int*)d_in[0];
    const int* seq_len  = (const int*)d_in[1];
    auto P = [&](int i) { return (const float*)d_in[i]; };
    const int HEAD_W = 44, HEAD_B = 45, ITEM_EMB = 46, LN_B = 47, LN_G = 48, POS_EMB = 49;

    // ---- workspace carve (all 256B-aligned) ----
    char* wp = (char*)d_ws;
    auto carve = [&](size_t bytes) { char* p = wp; wp += (bytes + 255) & ~(size_t)255; return p; };
    h16*   X    = (h16*)  carve((size_t)NTOK * 256 * 2);          // residual spike stream
    h16*   bufA = (h16*)  carve((size_t)(NTOK + 16) * 256 * 2);   // Sq / Sa / S1 (+pad rows)
    h16*   bufB = (h16*)  carve((size_t)(NTOK + 16) * 256 * 2);   // Sk / spill
    h16*   bufC = (h16*)  carve((size_t)(NTOK + 16) * 256 * 2);   // Sv
    float* G    = (float*)carve((size_t)NTOK * 256 * 4);          // f32 pre-LN scratch
    h16*   W16  = (h16*)  carve((size_t)12 * 65536 * 2);          // f16 weights

    // ---- convert weights to f16 (per block order: q,k,v,proj,fc1,fc2) ----
    for (int blk = 0; blk < 2; blk++) {
        const int base = 2 + 21 * blk;
        const float* srcs[6] = { P(base + 15), P(base + 8), P(base + 18),
                                 P(base + 11), P(base + 0), P(base + 4) };
        for (int m = 0; m < 6; m++)
            conv_f32_f16<<<256, 256, 0, stream>>>(srcs[m], W16 + (size_t)(blk * 6 + m) * 65536, 65536);
    }

    embed_kernel<<<BL, 256, 0, stream>>>(item_seq, P(ITEM_EMB), P(POS_EMB), P(LN_G), P(LN_B), X);

    const dim3 ggrid(NTOK / 128, 4);
    for (int blk = 0; blk < 2; blk++) {
        const int base = 2 + 21 * blk;
        const h16* Wq = W16 + (size_t)(blk * 6 + 0) * 65536;
        const h16* Wk = W16 + (size_t)(blk * 6 + 1) * 65536;
        const h16* Wv = W16 + (size_t)(blk * 6 + 2) * 65536;
        const h16* Wp = W16 + (size_t)(blk * 6 + 3) * 65536;
        const h16* W1 = W16 + (size_t)(blk * 6 + 4) * 65536;
        const h16* W2 = W16 + (size_t)(blk * 6 + 5) * 65536;

        // --- SSA ---
        gemm_xw_f16<<<ggrid, 256, 0, stream>>>(X, Wq, G);
        ln_lif_kernel<<<BL, 256, 0, stream>>>(G, nullptr, P(base + 17), P(base + 16), 1e-5f, 1.0f, bufA, nullptr);
        gemm_xw_f16<<<ggrid, 256, 0, stream>>>(X, Wk, G);
        ln_lif_kernel<<<BL, 256, 0, stream>>>(G, nullptr, P(base + 10), P(base + 9),  1e-5f, 1.0f, bufB, nullptr);
        gemm_xw_f16<<<ggrid, 256, 0, stream>>>(X, Wv, G);
        ln_lif_kernel<<<BL, 256, 0, stream>>>(G, nullptr, P(base + 20), P(base + 19), 1e-5f, 1.0f, bufC, nullptr);
        attention_kernel<<<dim3(13, 2, T_STEPS * BATCH), 256, 0, stream>>>(bufA, bufB, bufC, G);
        lif_kernel<<<BL, 256, 0, stream>>>(G, bufA, 0.5f);                     // attn_lif -> Sa
        gemm_xw_f16<<<ggrid, 256, 0, stream>>>(bufA, Wp, G);
        ln_lif_kernel<<<BL, 256, 0, stream>>>(G, P(base + 13), P(base + 14), P(base + 12),
                                              1e-5f, 1.0f, bufB, X);           // x += ssa(x)
        // --- MLP ---
        gemm_xw_f16<<<ggrid, 256, 0, stream>>>(X, W1, G);
        ln_lif_kernel<<<BL, 256, 0, stream>>>(G, P(base + 2), P(base + 3), P(base + 1),
                                              1e-5f, 1.0f, bufA, nullptr);
        gemm_xw_f16<<<ggrid, 256, 0, stream>>>(bufA, W2, G);
        ln_lif_kernel<<<BL, 256, 0, stream>>>(G, P(base + 6), P(base + 7), P(base + 5),
                                              1e-5f, 1.0f, bufB, X);           // x += mlp(x)
    }

    head_kernel<<<BATCH, 256, 0, stream>>>(X, seq_len, P(HEAD_W), P(HEAD_B), (float*)d_out);
}
